// AtomAttentionPairBias_66589172957468
// MI455X (gfx1250) — matrix operand order repaired and checked by reading, not compiled, not run on hardware
//
#include <hip/hip_runtime.h>
#include <hip/hip_bf16.h>

// ---------------------------------------------------------------------------
// AtomAttentionPairBias on gfx1250 (MI455X, wave32, WMMA).
// Dense math through v_wmma_f32_16x16x32_f16; GEMM weights staged per
// workgroup into LDS via global_load_async_to_lds_b128 (ASYNCcnt path).
// ---------------------------------------------------------------------------

typedef __attribute__((ext_vector_type(16))) _Float16     v16h;
typedef __attribute__((ext_vector_type(8)))  float        v8f;
typedef __attribute__((ext_vector_type(4)))  unsigned int v4u;

union H16 { v16h v; v4u u[2]; _Float16 h[16]; };

__device__ __forceinline__ v8f wmma_f16(v16h a, v16h b, v8f c) {
  return __builtin_amdgcn_wmma_f32_16x16x32_f16(false, a, false, b, (short)0, c,
                                                false, false);
}

__device__ __forceinline__ float sigmoidf(float x) {
  return 1.f / (1.f + __expf(-x));
}

// ---- problem constants ----
#define NN   2048
#define CC   128
#define HH   8
#define HD   16
#define NQ   32
#define NK   128
#define NB   64           // N / NQ
#define PAD  48           // (NK - NQ)/2
#define NPAD 2144         // N + 2*PAD

// ---------------------------------------------------------------------------
// Weight pack: w[k][n] (f32 row-major, KxN=128x128) -> wt[n][k] f16 (col-major)
// so the WMMA B operand is two k-contiguous 8-half loads per lane.
// ---------------------------------------------------------------------------
__global__ void pack_w_kernel(const float* __restrict__ w, _Float16* __restrict__ wt) {
  int tid = blockIdx.x * blockDim.x + threadIdx.x;   // 128*128
  int n = tid >> 7, k = tid & 127;
  wt[n * CC + k] = (_Float16)w[k * CC + n];
}

// ---------------------------------------------------------------------------
// Zero halos: kpad rows [0,48) & [2096,2144); vT positions [0,48) & [2096,2144)
// ---------------------------------------------------------------------------
__global__ void zero_halo_kernel(_Float16* __restrict__ kpad, _Float16* __restrict__ vT) {
  int tid = blockIdx.x * blockDim.x + threadIdx.x;   // 96*128
  int r = tid >> 7, c = tid & 127;
  int pr = (r < PAD) ? r : (NN + r);                 // 0..47 or 2096..2143
  kpad[(size_t)pr * CC + c] = (_Float16)0.f;
  vT[(size_t)c * NPAD + pr] = (_Float16)0.f;
}

// ---------------------------------------------------------------------------
// LayerNorm: a_ln = LN(x) (f32);  sn = LN(s)*adaln_s_w (f16).
// one block (128 threads) per row.
// ---------------------------------------------------------------------------
__global__ void ln_kernel(const float* __restrict__ x, const float* __restrict__ s,
                          const float* __restrict__ sw,
                          float* __restrict__ a_ln, _Float16* __restrict__ sn) {
  int row = blockIdx.x, c = threadIdx.x;
  size_t idx = (size_t)row * CC + c;
  float xv = x[idx], sv = s[idx];
  __shared__ float4 red[CC];
  red[c] = make_float4(xv, xv * xv, sv, sv * sv);
  __syncthreads();
  for (int off = 64; off > 0; off >>= 1) {
    if (c < off) {
      float4 a = red[c], b = red[c + off];
      red[c] = make_float4(a.x + b.x, a.y + b.y, a.z + b.z, a.w + b.w);
    }
    __syncthreads();
  }
  float4 t = red[0];
  float mx = t.x * (1.f / CC), ms = t.z * (1.f / CC);
  float vx = t.y * (1.f / CC) - mx * mx;
  float vs = t.w * (1.f / CC) - ms * ms;
  float aln = (xv - mx) * rsqrtf(vx + 1e-5f);
  float snv = (sv - ms) * rsqrtf(vs + 1e-5f) * sw[c];
  a_ln[idx] = aln;
  sn[idx] = (_Float16)snv;
}

// ---------------------------------------------------------------------------
// Generic 2048x128 @ 128x128 GEMM, one wave per 16-row tile, all 8 col tiles.
// The 32 KB weight matrix is async-copied ONCE per workgroup into LDS
// (global_load_async_to_lds_b128, ASYNCcnt), then all 4 waves feed their
// WMMA B operands from LDS (ds_load_b128) -> 4x less L2 traffic per block.
// A: f16 row-major.  WT: f16 [n][k].  Epilogue selected by `mode`:
//   0: o16 = acc + bias
//   1: o32 = acc
//   2: o16 = sigmoid(acc+bias)*aux1 + aux2          (AdaLN combine)
//   3: o16 = sigmoid(acc)*aux1                      (output gate)
//   4: o16 & o32 = acc
//   5: o32 = sigmoid(acc+bias)*aux1                 (final gated out)
//   6: transposed f16 store into vT [col][NPAD] at +PAD (V projection)
// ---------------------------------------------------------------------------
__global__ void gemm_kernel(const _Float16* __restrict__ A,
                            const _Float16* __restrict__ WT,
                            const float* __restrict__ bias, int mode,
                            const float* __restrict__ aux1,
                            const float* __restrict__ aux2,
                            float* __restrict__ o32, _Float16* __restrict__ o16) {
  __shared__ _Float16 wlds[CC * CC];                 // 32 KB staged weights

  // ---- async stage: 128 threads x 16 iters x 16B = 32 KB, GVS mode ----
  {
    unsigned lds_base = (unsigned)(size_t)(&wlds[0]);
#pragma unroll
    for (int it = 0; it < 16; ++it) {
      unsigned boff = (unsigned)((it * 128 + threadIdx.x) * 16);
      asm volatile("global_load_async_to_lds_b128 %0, %1, %2"
                   :
                   : "v"(lds_base + boff), "v"(boff), "s"(WT)
                   : "memory");
    }
    asm volatile("s_wait_asynccnt 0x0" ::: "memory");
  }
  __syncthreads();

  int wid = threadIdx.x >> 5, lane = threadIdx.x & 31;
  int tile = blockIdx.x * 4 + wid;                   // 128 row tiles
  int lo = lane & 15;
  bool hi = lane >= 16;
  int row = tile * 16 + lo;

  // A operand: 4 chunks of K=32, held in registers, reused for all 8 col tiles
  H16 Areg[4];
  const _Float16* ap0 = A + (size_t)row * CC + (hi ? 8 : 0);
#pragma unroll
  for (int kc = 0; kc < 4; ++kc) {
    Areg[kc].u[0] = *(const v4u*)(ap0 + kc * 32);
    Areg[kc].u[1] = *(const v4u*)(ap0 + kc * 32 + 16);
  }

#pragma unroll
  for (int nt = 0; nt < 8; ++nt) {
    int col = nt * 16 + lo;
    const _Float16* wp = wlds + (size_t)col * CC + (hi ? 8 : 0);
    v8f acc = {};
#pragma unroll
    for (int kc = 0; kc < 4; ++kc) {
      H16 B;
      B.u[0] = *(const v4u*)(wp + kc * 32);
      B.u[1] = *(const v4u*)(wp + kc * 32 + 16);
      acc = wmma_f16(Areg[kc].v, B.v, acc);
    }
    float bcol = bias ? bias[col] : 0.f;
    if (mode == 6) {
      union { v4u u; _Float16 h[8]; } pk;
#pragma unroll
      for (int r = 0; r < 8; ++r) pk.h[r] = (_Float16)acc[r];
      _Float16* op = o16 + (size_t)col * NPAD + PAD + tile * 16 + (hi ? 8 : 0);
      *(v4u*)op = pk.u;
    } else {
#pragma unroll
      for (int r = 0; r < 8; ++r) {
        int m = tile * 16 + r + (hi ? 8 : 0);
        size_t idx = (size_t)m * CC + col;
        float v = acc[r] + bcol;
        switch (mode) {
          case 0: o16[idx] = (_Float16)v; break;
          case 1: o32[idx] = v; break;
          case 2: o16[idx] = (_Float16)(sigmoidf(v) * aux1[idx] + aux2[idx]); break;
          case 3: o16[idx] = (_Float16)(sigmoidf(acc[r]) * aux1[idx]); break;
          case 4: o32[idx] = v; o16[idx] = (_Float16)v; break;
          case 5: o32[idx] = sigmoidf(v) * aux1[idx]; break;
          default: break;
        }
      }
    }
  }
}

// ---------------------------------------------------------------------------
// Pair bias: only the windowed slices of pair (16.8 MB of 268 MB) are read.
// One thread per (q_global, k_window).  bias layout [blk][h][ql][kw] f32.
// ---------------------------------------------------------------------------
__global__ void pair_kernel(const float* __restrict__ pair,
                            const float* __restrict__ lnw,
                            const float* __restrict__ lnb,
                            const float* __restrict__ projw,
                            float* __restrict__ bias) {
  __shared__ float sp[16 * 8], slw[16], slb[16];
  if (threadIdx.x < 128) sp[threadIdx.x] = projw[threadIdx.x];
  if (threadIdx.x < 16) { slw[threadIdx.x] = lnw[threadIdx.x]; slb[threadIdx.x] = lnb[threadIdx.x]; }
  __syncthreads();

  int tid = blockIdx.x * blockDim.x + threadIdx.x;   // 2048*128
  int qg = tid >> 7, kw = tid & 127;
  int blk = qg >> 5, ql = qg & 31;
  int kg = blk * NQ + kw - PAD;

  float out[HH];
#pragma unroll
  for (int h = 0; h < HH; ++h) out[h] = 0.f;

  if (kg >= 0 && kg < NN) {
    const float4* p4 = (const float4*)(pair + ((size_t)qg * NN + kg) * 16);
    float v[16];
#pragma unroll
    for (int j = 0; j < 4; ++j) {
      float4 t = p4[j];
      v[4 * j + 0] = t.x; v[4 * j + 1] = t.y; v[4 * j + 2] = t.z; v[4 * j + 3] = t.w;
    }
    float m = 0.f, ss = 0.f;
#pragma unroll
    for (int j = 0; j < 16; ++j) { m += v[j]; ss += v[j] * v[j]; }
    m *= (1.f / 16.f);
    float var = ss * (1.f / 16.f) - m * m;
    float rs = rsqrtf(var + 1e-5f);
#pragma unroll
    for (int j = 0; j < 16; ++j) {
      float lnv = (v[j] - m) * rs * slw[j] + slb[j];
#pragma unroll
      for (int h = 0; h < HH; ++h) out[h] += lnv * sp[j * HH + h];
    }
  }
  float* bp = bias + (((size_t)blk * HH) * NQ + ql) * NK + kw;
#pragma unroll
  for (int h = 0; h < HH; ++h) bp[(size_t)h * NQ * NK] = out[h];
}

// ---------------------------------------------------------------------------
// Windowed attention.  One wave per (blk, head, 16-row q subtile).
// logits: WMMA f16 with hd=16 zero-padded to K=32; softmax in registers
// (rows live in 16-lane halves of the C/D layout, __shfl_xor reductions);
// P re-swizzled via LDS into A-operand layout; PV via WMMA with vT B-operand.
// ---------------------------------------------------------------------------
__global__ void attn_kernel(const _Float16* __restrict__ q,
                            const _Float16* __restrict__ kpad,
                            const _Float16* __restrict__ vT,
                            const float* __restrict__ bias,
                            float* __restrict__ o32, _Float16* __restrict__ o16) {
  __shared__ _Float16 pbuf[4][16 * NK];
  int wid = threadIdx.x >> 5, lane = threadIdx.x & 31;
  int task = blockIdx.x * 4 + wid;                   // 1024 tasks
  int mt = task & 1;
  int h = (task >> 1) & 7;
  int blk = task >> 4;
  int lo = lane & 15;
  bool hi = lane >= 16;
  _Float16* lp = pbuf[wid];

  v4u vz = (v4u)0u;

  // A operand for Q: rows m = mt*16 + lo, K = head dim (16 real + 16 zero pad)
  H16 Aq;
  {
    const _Float16* qp = q + (size_t)(blk * NQ + mt * 16 + lo) * CC + h * HD + (hi ? 8 : 0);
    Aq.u[0] = *(const v4u*)qp;
    Aq.u[1] = vz;                                    // K = 16..31 zero pad
  }

  const float* bp = bias + (((size_t)blk * HH + h) * NQ + mt * 16) * NK;

  float acc[8][8];                                   // [ktile][r]
#pragma unroll
  for (int kt = 0; kt < 8; ++kt) {
    H16 Bk;
    {
      const _Float16* kp = kpad + (size_t)(blk * NQ + kt * 16 + lo) * CC + h * HD + (hi ? 8 : 0);
      Bk.u[0] = *(const v4u*)kp;
      Bk.u[1] = vz;                                  // K = 16..31 zero pad
    }
    v8f c = {};
    c = wmma_f16(Aq.v, Bk.v, c);
#pragma unroll
    for (int r = 0; r < 8; ++r) {
      int m = r + (hi ? 8 : 0);
      acc[kt][r] = c[r] * 0.25f + bp[(size_t)m * NK + kt * 16 + lo];
    }
  }

  // ---- softmax over 128 keys per row (rows confined to 16-lane halves) ----
  float rsum[8];
#pragma unroll
  for (int r = 0; r < 8; ++r) {
    float mx = acc[0][r];
#pragma unroll
    for (int kt = 1; kt < 8; ++kt) mx = fmaxf(mx, acc[kt][r]);
#pragma unroll
    for (int msk = 1; msk < 16; msk <<= 1) mx = fmaxf(mx, __shfl_xor(mx, msk, 32));
    float sm = 0.f;
#pragma unroll
    for (int kt = 0; kt < 8; ++kt) { acc[kt][r] = __expf(acc[kt][r] - mx); sm += acc[kt][r]; }
#pragma unroll
    for (int msk = 1; msk < 16; msk <<= 1) sm += __shfl_xor(sm, msk, 32);
    rsum[r] = 1.f / sm;
  }

  // stage P (f16) into LDS [m][k] for A-operand re-swizzle
#pragma unroll
  for (int kt = 0; kt < 8; ++kt)
#pragma unroll
    for (int r = 0; r < 8; ++r) {
      int m = r + (hi ? 8 : 0);
      lp[m * NK + kt * 16 + lo] = (_Float16)(acc[kt][r] * rsum[r]);
    }
  // same-wave DS ops are in-order: no barrier needed (private LDS slice)

  // ---- PV: K = 128 keys in 4 chunks of 32 ----
  v8f oacc = {};
#pragma unroll
  for (int kc = 0; kc < 4; ++kc) {
    H16 Ap;
    const _Float16* ap = lp + lo * NK + kc * 32 + (hi ? 8 : 0);
    Ap.u[0] = *(const v4u*)ap;
    Ap.u[1] = *(const v4u*)(ap + 16);
    H16 Bv;
    const _Float16* vp = vT + (size_t)(h * HD + lo) * NPAD + blk * NQ + kc * 32 + (hi ? 8 : 0);
    Bv.u[0] = *(const v4u*)vp;
    Bv.u[1] = *(const v4u*)(vp + 16);
    oacc = wmma_f16(Ap.v, Bv.v, oacc);
  }
#pragma unroll
  for (int r = 0; r < 8; ++r) {
    int m = blk * NQ + mt * 16 + r + (hi ? 8 : 0);
    size_t idx = (size_t)m * CC + h * HD + lo;
    o32[idx] = oacc[r];
    o16[idx] = (_Float16)oacc[r];
  }
}

// ---------------------------------------------------------------------------
// Host-side orchestration (graph-capture safe: only kernel launches on stream)
// ---------------------------------------------------------------------------
extern "C" void kernel_launch(void* const* d_in, const int* in_sizes, int n_in,
                              void* d_out, int out_size, void* d_ws, size_t ws_size,
                              hipStream_t stream) {
  const float* x            = (const float*)d_in[0];
  const float* s            = (const float*)d_in[1];
  const float* pair         = (const float*)d_in[2];
  const float* adaln_s_w    = (const float*)d_in[3];
  const float* adaln_gate_w = (const float*)d_in[4];
  const float* adaln_gate_b = (const float*)d_in[5];
  const float* adaln_skip_w = (const float*)d_in[6];
  const float* q_w          = (const float*)d_in[7];
  const float* q_b          = (const float*)d_in[8];
  const float* k_w          = (const float*)d_in[9];
  const float* v_w          = (const float*)d_in[10];
  const float* pair_ln_w    = (const float*)d_in[11];
  const float* pair_ln_b    = (const float*)d_in[12];
  const float* pair_proj_w  = (const float*)d_in[13];
  const float* gate_w       = (const float*)d_in[14];
  const float* attn_proj_w  = (const float*)d_in[15];
  const float* out_w        = (const float*)d_in[16];
  const float* out_b        = (const float*)d_in[17];

  char* w = (char*)d_ws;
  const size_t WSZ = (size_t)CC * CC * sizeof(_Float16);        // 32768
  _Float16* WT0 = (_Float16*)(w + 0 * WSZ);   // adaln_gate_w^T
  _Float16* WT1 = (_Float16*)(w + 1 * WSZ);   // adaln_skip_w^T
  _Float16* WT2 = (_Float16*)(w + 2 * WSZ);   // q_w^T
  _Float16* WT3 = (_Float16*)(w + 3 * WSZ);   // k_w^T
  _Float16* WT4 = (_Float16*)(w + 4 * WSZ);   // v_w^T
  _Float16* WT5 = (_Float16*)(w + 5 * WSZ);   // gate_w^T
  _Float16* WT6 = (_Float16*)(w + 6 * WSZ);   // attn_proj_w^T
  _Float16* WT7 = (_Float16*)(w + 7 * WSZ);   // out_w^T

  size_t off = 8 * WSZ;                                          // 262144
  float*     a_ln = (float*)(w + off);     off += (size_t)NN * CC * 4;
  _Float16*  sn   = (_Float16*)(w + off);  off += (size_t)NN * CC * 2;
  float*     Sbuf = (float*)(w + off);     off += (size_t)NN * CC * 4;
  _Float16*  a16  = (_Float16*)(w + off);  off += (size_t)NN * CC * 2;
  _Float16*  q16  = (_Float16*)(w + off);  off += (size_t)NN * CC * 2;
  _Float16*  kpad = (_Float16*)(w + off);  off += (size_t)NPAD * CC * 2;
  _Float16*  vT   = (_Float16*)(w + off);  off += (size_t)CC * NPAD * 2;
  float*     bias = (float*)(w + off);     off += (size_t)NB * HH * NQ * NK * 4;
  float*     o32  = (float*)(w + off);     off += (size_t)NN * CC * 4;
  _Float16*  o16  = (_Float16*)(w + off);  off += (size_t)NN * CC * 2;
  _Float16*  g16  = (_Float16*)(w + off);  off += (size_t)NN * CC * 2;
  _Float16*  u16  = (_Float16*)(w + off);  off += (size_t)NN * CC * 2;
  float*     u32  = (float*)(w + off);     off += (size_t)NN * CC * 4;
  (void)ws_size; (void)n_in; (void)in_sizes; (void)out_size;

  // 0) pack weights + zero halos
  pack_w_kernel<<<128, 128, 0, stream>>>(adaln_gate_w, WT0);
  pack_w_kernel<<<128, 128, 0, stream>>>(adaln_skip_w, WT1);
  pack_w_kernel<<<128, 128, 0, stream>>>(q_w, WT2);
  pack_w_kernel<<<128, 128, 0, stream>>>(k_w, WT3);
  pack_w_kernel<<<128, 128, 0, stream>>>(v_w, WT4);
  pack_w_kernel<<<128, 128, 0, stream>>>(gate_w, WT5);
  pack_w_kernel<<<128, 128, 0, stream>>>(attn_proj_w, WT6);
  pack_w_kernel<<<128, 128, 0, stream>>>(out_w, WT7);
  zero_halo_kernel<<<48, 256, 0, stream>>>(kpad, vT);

  // 1) LayerNorms
  ln_kernel<<<NN, 128, 0, stream>>>(x, s, adaln_s_w, a_ln, sn);

  // 2) AdaLN: S = sn@skip;  a = sigmoid(sn@gate + b)*a_ln + S
  gemm_kernel<<<32, 128, 0, stream>>>(sn, WT1, nullptr, 1, nullptr, nullptr, Sbuf, nullptr);
  gemm_kernel<<<32, 128, 0, stream>>>(sn, WT0, adaln_gate_b, 2, a_ln, Sbuf, nullptr, a16);

  // 3) Q/K/V projections (K into padded buffer, V transposed+padded)
  gemm_kernel<<<32, 128, 0, stream>>>(a16, WT2, q_b, 0, nullptr, nullptr, nullptr, q16);
  gemm_kernel<<<32, 128, 0, stream>>>(a16, WT3, nullptr, 0, nullptr, nullptr, nullptr,
                                      kpad + (size_t)PAD * CC);
  gemm_kernel<<<32, 128, 0, stream>>>(a16, WT4, nullptr, 6, nullptr, nullptr, nullptr, vT);

  // 4) pair bias (windowed slices only)
  pair_kernel<<<1024, 256, 0, stream>>>(pair, pair_ln_w, pair_ln_b, pair_proj_w, bias);

  // 5) windowed attention
  attn_kernel<<<256, 128, 0, stream>>>(q16, kpad, vT, bias, o32, o16);

  // 6) gated output path
  gemm_kernel<<<32, 128, 0, stream>>>(o16, WT5, nullptr, 3, o32, nullptr, nullptr, g16);
  gemm_kernel<<<32, 128, 0, stream>>>(g16, WT6, nullptr, 4, nullptr, nullptr, u32, u16);
  gemm_kernel<<<32, 128, 0, stream>>>(u16, WT7, out_b, 5, u32, nullptr, (float*)d_out, nullptr);
}
